// AttGraphConvLayer_53541062312452
// MI455X (gfx1250) — matrix operand order repaired and checked by reading, not compiled, not run on hardware
//
#include <hip/hip_runtime.h>
#include <hip/hip_bf16.h>

// Problem constants (match reference)
constexpr int BB = 4;
constexpr int EE = 8192;
constexpr int NN = 1024;
constexpr int FF = 512;
constexpr int DD = 512;
#define ALPHA 0.2f
#define NEGC  (-1e9f)

typedef __attribute__((ext_vector_type(16))) __bf16 v16bf;
typedef __attribute__((ext_vector_type(8)))  float  v8f;

// ---------------- wave32 reductions via ds_swizzle (xor butterflies) -------
__device__ __forceinline__ float wred_sum(float x) {
  x += __int_as_float(__builtin_amdgcn_ds_swizzle(__float_as_int(x), 0x041f)); // xor 1
  x += __int_as_float(__builtin_amdgcn_ds_swizzle(__float_as_int(x), 0x081f)); // xor 2
  x += __int_as_float(__builtin_amdgcn_ds_swizzle(__float_as_int(x), 0x101f)); // xor 4
  x += __int_as_float(__builtin_amdgcn_ds_swizzle(__float_as_int(x), 0x201f)); // xor 8
  x += __int_as_float(__builtin_amdgcn_ds_swizzle(__float_as_int(x), 0x401f)); // xor 16
  return x;
}
__device__ __forceinline__ float wred_max(float x) {
  x = fmaxf(x, __int_as_float(__builtin_amdgcn_ds_swizzle(__float_as_int(x), 0x041f)));
  x = fmaxf(x, __int_as_float(__builtin_amdgcn_ds_swizzle(__float_as_int(x), 0x081f)));
  x = fmaxf(x, __int_as_float(__builtin_amdgcn_ds_swizzle(__float_as_int(x), 0x101f)));
  x = fmaxf(x, __int_as_float(__builtin_amdgcn_ds_swizzle(__float_as_int(x), 0x201f)));
  x = fmaxf(x, __int_as_float(__builtin_amdgcn_ds_swizzle(__float_as_int(x), 0x401f)));
  return x;
}
__device__ __forceinline__ int wred_imax(int x) {
  int y;
  y = __builtin_amdgcn_ds_swizzle(x, 0x041f); x = (x > y) ? x : y;
  y = __builtin_amdgcn_ds_swizzle(x, 0x081f); x = (x > y) ? x : y;
  y = __builtin_amdgcn_ds_swizzle(x, 0x101f); x = (x > y) ? x : y;
  y = __builtin_amdgcn_ds_swizzle(x, 0x201f); x = (x > y) ? x : y;
  y = __builtin_amdgcn_ds_swizzle(x, 0x401f); x = (x > y) ? x : y;
  return x;
}

// ---------------- small helper kernels -------------------------------------
__global__ void k_zero4(float4* __restrict__ p, int n4) {
  int i = blockIdx.x * blockDim.x + threadIdx.x;
  if (i < n4) p[i] = make_float4(0.f, 0.f, 0.f, 0.f);
}

// Generic fp32 -> bf16 hi/lo split (row-major, same layout)
__global__ void k_split(const float* __restrict__ x,
                        __bf16* __restrict__ xh, __bf16* __restrict__ xl, int n) {
  int i = blockIdx.x * blockDim.x + threadIdx.x;
  if (i >= n) return;
  float v = x[i];
  __bf16 h = (__bf16)v;
  xh[i] = h;
  xl[i] = (__bf16)(v - (float)h);
}

// Recover src/dst indices from one-hot incidence matrices: one wave per edge.
__global__ void k_extract_idx(const float* __restrict__ Cmat,
                              const float* __restrict__ Nmat,
                              int* __restrict__ src, int* __restrict__ dst) {
  int wave = (blockIdx.x * blockDim.x + threadIdx.x) >> 5;
  int lane = threadIdx.x & 31;
  if (wave >= BB * EE) return;
  const float* rc = Cmat + (size_t)wave * NN;
  const float* rn = Nmat + (size_t)wave * NN;
  int ic = -1, id = -1;
  for (int j = lane; j < NN; j += 32) {
    if (rc[j] > 0.5f) ic = j;
    if (rn[j] > 0.5f) id = j;
  }
  ic = wred_imax(ic);
  id = wred_imax(id);
  if (lane == 0) { src[wave] = ic; dst[wave] = id; }
}

// w [F,D] fp32 -> wT hi/lo bf16 [D][F] (K contiguous for WMMA B operand)
__global__ void k_wt_split(const float* __restrict__ w,
                           __bf16* __restrict__ wth, __bf16* __restrict__ wtl) {
  int i = blockIdx.x * blockDim.x + threadIdx.x;
  if (i >= FF * DD) return;
  int f = i / DD, d = i - f * DD;
  float x = w[i];
  __bf16 h = (__bf16)x;
  __bf16 l = (__bf16)(x - (float)h);
  wth[(size_t)d * FF + f] = h;
  wtl[(size_t)d * FF + f] = l;
}

// z [B][N][D] fp32 -> zT hi/lo bf16 [B][D][N]
__global__ void k_zt_split(const float* __restrict__ Z,
                           __bf16* __restrict__ zth, __bf16* __restrict__ ztl) {
  int i = blockIdx.x * blockDim.x + threadIdx.x;
  if (i >= BB * NN * DD) return;
  int d = i % DD;
  int n = (i / DD) % NN;
  int b = i / (DD * NN);
  float x = Z[i];
  __bf16 h = (__bf16)x;
  __bf16 l = (__bf16)(x - (float)h);
  size_t o = ((size_t)b * DD + d) * NN + n;
  zth[o] = h;
  ztl[o] = l;
}

// s1[b,n] = z[b,n,:]·attn[0:D], s2[b,n] = z[b,n,:]·attn[D:2D]; wave per node.
__global__ void k_s1s2(const float* __restrict__ Z, const float* __restrict__ attn,
                       float* __restrict__ s1, float* __restrict__ s2) {
  int wave = (blockIdx.x * blockDim.x + threadIdx.x) >> 5;
  int lane = threadIdx.x & 31;
  if (wave >= BB * NN) return;
  const float* zr = Z + (size_t)wave * DD;
  float a1 = 0.f, a2 = 0.f;
  for (int d = lane; d < DD; d += 32) {
    float zv = zr[d];
    a1 += zv * attn[d];
    a2 += zv * attn[DD + d];
  }
  a1 = wred_sum(a1);
  a2 = wred_sum(a2);
  if (lane == 0) { s1[wave] = a1; s2[wave] = a2; }
}

// att[b,e] = leaky_relu(s1[b,src]+s2[b,dst])
__global__ void k_att(const int* __restrict__ src, const int* __restrict__ dst,
                      const float* __restrict__ s1, const float* __restrict__ s2,
                      float* __restrict__ att) {
  int i = blockIdx.x * blockDim.x + threadIdx.x;
  if (i >= BB * EE) return;
  int b = i / EE;
  float v = s1[b * NN + src[i]] + s2[b * NN + dst[i]];
  att[i] = (v > 0.f) ? v : ALPHA * v;
}

// Scatter per-edge score and adjacency count into dense [B,N,N]
__global__ void k_scatter(const int* __restrict__ src, const int* __restrict__ dst,
                          const float* __restrict__ att,
                          float* __restrict__ scores, float* __restrict__ adj) {
  int i = blockIdx.x * blockDim.x + threadIdx.x;
  if (i >= BB * EE) return;
  int b = i / EE;
  size_t off = ((size_t)b * NN + src[i]) * NN + dst[i];
  atomicAdd(scores + off, att[i]);
  atomicAdd(adj + off, 1.0f);
}

// Masked softmax per row (keras Softmax(mask=adj) semantics), times adj.
// Emits bf16 hi/lo adj_norm (A operand for GEMM2). One wave per row.
__global__ void k_softmax(const float* __restrict__ scores, const float* __restrict__ adj,
                          __bf16* __restrict__ anh, __bf16* __restrict__ anl) {
  int wave = (blockIdx.x * blockDim.x + threadIdx.x) >> 5;
  int lane = threadIdx.x & 31;
  if (wave >= BB * NN) return;
  const float* srow = scores + (size_t)wave * NN;
  const float* arow = adj + (size_t)wave * NN;
  __bf16* hrow = anh + (size_t)wave * NN;
  __bf16* lrow = anl + (size_t)wave * NN;
  float lg[NN / 32];
  float lmax = -__builtin_inff();
  #pragma unroll
  for (int t = 0; t < NN / 32; ++t) {
    int j = lane + t * 32;
    float a = arow[j];
    float l = srow[j] + (1.0f - a) * NEGC;
    lg[t] = l;
    lmax = fmaxf(lmax, l);
  }
  lmax = wred_max(lmax);
  float lsum = 0.f;
  #pragma unroll
  for (int t = 0; t < NN / 32; ++t) {
    float e = __expf(lg[t] - lmax);
    lg[t] = e;
    lsum += e;
  }
  lsum = wred_sum(lsum);
  float inv = 1.0f / lsum;
  #pragma unroll
  for (int t = 0; t < NN / 32; ++t) {
    int j = lane + t * 32;
    float v = lg[t] * inv * arow[j];
    __bf16 h = (__bf16)v;
    hrow[j] = h;
    lrow[j] = (__bf16)(v - (float)h);
  }
}

// ---------------- bf16x3 WMMA GEMM with async-LDS B staging ----------------
// C[M,Nc] = (Ah+Al)[M,K] @ (Bth+Btl)[Nc,K]^T  (3-term bf16 product, fp32 acc)
// A: pre-split bf16 hi/lo, row-major, K contiguous (direct b128 fragment loads)
// B: staged per K-step into LDS via global_load_async_to_lds_b128,
//    fragments read as ds_load_b128. Block 256 thr = 8 waves (4M x 2N),
//    wave tile 32x32, block tile 128x64.
__global__ void k_gemm_bf16x3(const __bf16* __restrict__ Ah,
                              const __bf16* __restrict__ Al,
                              const __bf16* __restrict__ Bth,
                              const __bf16* __restrict__ Btl,
                              float* __restrict__ C,
                              int Ncols, int K,
                              long long strideA, long long strideB, long long strideC,
                              int doLeaky) {
  __shared__ __align__(16) char sbuf[8192];  // [0,4096): B_hi tile, [4096,8192): B_lo

  const int tid  = threadIdx.x;
  const int lane = tid & 31;
  const int wave = tid >> 5;
  const int wm = wave & 3;
  const int wn = wave >> 2;
  const int half = lane >> 4;
  const int ml = lane & 15;
  const int b = blockIdx.z;

  Ah  += (size_t)b * strideA;
  Al  += (size_t)b * strideA;
  Bth += (size_t)b * strideB;
  Btl += (size_t)b * strideB;
  C   += (size_t)b * strideC;

  const int row0 = blockIdx.x * 128 + wm * 32;
  const int colB = blockIdx.y * 64;           // block's 64 output columns
  const int col0 = colB + wn * 32;

  // Async staging assignment: thread t copies 16B chunk (t&3) of column (t>>2).
  const int scol   = tid >> 2;                // 0..63
  const int schunk = tid & 3;                 // 0..3 (16B each -> 64B = 32 k-values)
  const unsigned lbase  = (unsigned)(unsigned long long)(void*)sbuf;
  const unsigned lds_hi = lbase + (unsigned)(scol * 64 + schunk * 16);
  const unsigned lds_lo = lds_hi + 4096u;

  v8f acc[2][2] = {};

  for (int k0 = 0; k0 < K; k0 += 32) {
    // ---- stage B tile (64 cols x 32 k, hi+lo) into LDS asynchronously ----
    {
      const char* gh = (const char*)(Bth + (size_t)(colB + scol) * K + k0) + schunk * 16;
      const char* gl = (const char*)(Btl + (size_t)(colB + scol) * K + k0) + schunk * 16;
      asm volatile("global_load_async_to_lds_b128 %0, %1, off"
                   :: "v"(lds_hi), "v"(gh) : "memory");
      asm volatile("global_load_async_to_lds_b128 %0, %1, off"
                   :: "v"(lds_lo), "v"(gl) : "memory");
      asm volatile("s_wait_asynccnt 0" ::: "memory");
    }
    __syncthreads();

    // ---- A fragments: two contiguous 16B chunks per lane (pre-split bf16) ---
    v16bf ah[2], al[2];
    #pragma unroll
    for (int mi = 0; mi < 2; ++mi) {
      const __bf16* pah = Ah + (size_t)(row0 + mi * 16 + ml) * K + k0 + 8 * half;
      const __bf16* pal = Al + (size_t)(row0 + mi * 16 + ml) * K + k0 + 8 * half;
      __builtin_memcpy((char*)&ah[mi],      pah,      16);  // k: 8h .. 8h+7
      __builtin_memcpy((char*)&ah[mi] + 16, pah + 16, 16);  // k: 16+8h .. 16+8h+7
      __builtin_memcpy((char*)&al[mi],      pal,      16);
      __builtin_memcpy((char*)&al[mi] + 16, pal + 16, 16);
    }

    // ---- B fragments from LDS: two 16B chunks per lane, 32B apart ----------------
    v16bf bh[2], bl[2];
    #pragma unroll
    for (int ni = 0; ni < 2; ++ni) {
      const char* bp = sbuf + (wn * 32 + ni * 16 + ml) * 64 + 16 * half;
      __builtin_memcpy((char*)&bh[ni],      bp,        16);  // k: 8h .. 8h+7
      __builtin_memcpy((char*)&bh[ni] + 16, bp + 32,   16);  // k: 16+8h .. 16+8h+7
      __builtin_memcpy((char*)&bl[ni],      bp + 4096, 16);
      __builtin_memcpy((char*)&bl[ni] + 16, bp + 4128, 16);
    }

    // ---- bf16x3: ah*bh + ah*bl + al*bh ------------------------------------
    #pragma unroll
    for (int mi = 0; mi < 2; ++mi) {
      #pragma unroll
      for (int ni = 0; ni < 2; ++ni) {
        acc[mi][ni] = __builtin_amdgcn_wmma_f32_16x16x32_bf16(
            false, ah[mi], false, bh[ni], (short)0, acc[mi][ni], false, false);
        acc[mi][ni] = __builtin_amdgcn_wmma_f32_16x16x32_bf16(
            false, ah[mi], false, bl[ni], (short)0, acc[mi][ni], false, false);
        acc[mi][ni] = __builtin_amdgcn_wmma_f32_16x16x32_bf16(
            false, al[mi], false, bh[ni], (short)0, acc[mi][ni], false, false);
      }
    }
    __syncthreads();  // protect LDS tile before next-iteration overwrite
  }

  // Epilogue: C/D layout -> row = r + 8*half, col = ml
  #pragma unroll
  for (int mi = 0; mi < 2; ++mi) {
    #pragma unroll
    for (int ni = 0; ni < 2; ++ni) {
      #pragma unroll
      for (int r = 0; r < 8; ++r) {
        const int rr = row0 + mi * 16 + r + 8 * half;
        const int cc = col0 + ni * 16 + ml;
        float v = acc[mi][ni][r];
        if (doLeaky) v = (v > 0.f) ? v : ALPHA * v;
        C[(size_t)rr * Ncols + cc] = v;
      }
    }
  }
}

// ---------------- workspace layout (bytes, all 256B-aligned sizes) ---------
constexpr size_t SZ_Z   = (size_t)BB * NN * DD * 4;   //  8 MiB  z fp32
constexpr size_t SZ_SC  = (size_t)BB * NN * NN * 4;   // 16 MiB  scores
constexpr size_t SZ_ADJ = SZ_SC;                      // 16 MiB  adjacency counts
constexpr size_t SZ_S   = (size_t)BB * NN * 4;        // 16 KiB  s1/s2
constexpr size_t SZ_IDX = (size_t)BB * EE * 4;        // 128 KiB src/dst/att
constexpr size_t SZ_WT  = (size_t)FF * DD * 2;        // 512 KiB wT hi/lo bf16
constexpr size_t SZ_ZT  = (size_t)BB * DD * NN * 2;   //  4 MiB  zT hi/lo bf16
constexpr size_t SZ_NH  = (size_t)BB * NN * FF * 2;   //  4 MiB  nodes hi/lo bf16
constexpr size_t SZ_AN  = (size_t)BB * NN * NN * 2;   //  8 MiB  adj_norm hi/lo bf16

constexpr size_t OFF_Z   = 0;
constexpr size_t OFF_SC  = OFF_Z + SZ_Z;
constexpr size_t OFF_ADJ = OFF_SC + SZ_SC;
constexpr size_t OFF_S1  = OFF_ADJ + SZ_ADJ;
constexpr size_t OFF_S2  = OFF_S1 + SZ_S;
constexpr size_t OFF_SRC = OFF_S2 + SZ_S;
constexpr size_t OFF_DST = OFF_SRC + SZ_IDX;
constexpr size_t OFF_ATT = OFF_DST + SZ_IDX;
constexpr size_t OFF_WTH = OFF_ATT + SZ_IDX;
constexpr size_t OFF_WTL = OFF_WTH + SZ_WT;
constexpr size_t OFF_ZTH = OFF_WTL + SZ_WT;
constexpr size_t OFF_ZTL = OFF_ZTH + SZ_ZT;
constexpr size_t OFF_NH  = OFF_ZTL + SZ_ZT;
constexpr size_t OFF_NL  = OFF_NH + SZ_NH;
constexpr size_t OFF_ANH = OFF_NL + SZ_NH;
constexpr size_t OFF_ANL = OFF_ANH + SZ_AN;

extern "C" void kernel_launch(void* const* d_in, const int* in_sizes, int n_in,
                              void* d_out, int out_size, void* d_ws, size_t ws_size,
                              hipStream_t stream) {
  (void)in_sizes; (void)n_in; (void)out_size; (void)ws_size;

  const float* nodes = (const float*)d_in[0];   // [B,N,F]
  const float* Cmat  = (const float*)d_in[1];   // [B,E,N]
  const float* Nmat  = (const float*)d_in[2];   // [B,E,N]
  // d_in[3] = mask (unused by reference call path)
  const float* w     = (const float*)d_in[4];   // [F,D]
  const float* attn  = (const float*)d_in[5];   // [2D,1]

  char* ws = (char*)d_ws;
  float*  Z      = (float*)(ws + OFF_Z);
  float*  SCORES = (float*)(ws + OFF_SC);
  float*  ADJ    = (float*)(ws + OFF_ADJ);
  float*  S1     = (float*)(ws + OFF_S1);
  float*  S2     = (float*)(ws + OFF_S2);
  int*    SRC    = (int*)(ws + OFF_SRC);
  int*    DST    = (int*)(ws + OFF_DST);
  float*  ATT    = (float*)(ws + OFF_ATT);
  __bf16* WTH    = (__bf16*)(ws + OFF_WTH);
  __bf16* WTL    = (__bf16*)(ws + OFF_WTL);
  __bf16* ZTH    = (__bf16*)(ws + OFF_ZTH);
  __bf16* ZTL    = (__bf16*)(ws + OFF_ZTL);
  __bf16* NH     = (__bf16*)(ws + OFF_NH);
  __bf16* NL     = (__bf16*)(ws + OFF_NL);
  __bf16* ANH    = (__bf16*)(ws + OFF_ANH);
  __bf16* ANL    = (__bf16*)(ws + OFF_ANL);

  float* OUT = (float*)d_out;                   // [B,N,D] fp32

  // 1) operand prep: w -> wT hi/lo; nodes -> hi/lo (row-major)
  k_wt_split<<<(FF * DD + 255) / 256, 256, 0, stream>>>(w, WTH, WTL);
  k_split<<<(BB * NN * FF + 255) / 256, 256, 0, stream>>>(nodes, NH, NL, BB * NN * FF);

  // 2) recover src/dst from one-hot incidence (wave per edge)
  k_extract_idx<<<(BB * EE * 32) / 256, 256, 0, stream>>>(Cmat, Nmat, SRC, DST);

  // 3) zero dense scores + adjacency
  k_zero4<<<(BB * NN * NN / 4 + 255) / 256, 256, 0, stream>>>((float4*)SCORES, BB * NN * NN / 4);
  k_zero4<<<(BB * NN * NN / 4 + 255) / 256, 256, 0, stream>>>((float4*)ADJ, BB * NN * NN / 4);

  // 4) GEMM1: Z[B*N, D] = nodes[B*N, F] @ w[F, D]   (bf16x3 WMMA, async B)
  {
    dim3 grid((BB * NN) / 128, DD / 64, 1);
    k_gemm_bf16x3<<<grid, 256, 0, stream>>>(NH, NL, WTH, WTL, Z,
                                            DD, FF, 0, 0, 0, /*leaky=*/0);
  }

  // 5) Z -> zT hi/lo bf16 per batch
  k_zt_split<<<(BB * NN * DD + 255) / 256, 256, 0, stream>>>(Z, ZTH, ZTL);

  // 6) per-node attention partials
  k_s1s2<<<(BB * NN * 32) / 256, 256, 0, stream>>>(Z, attn, S1, S2);

  // 7) per-edge attention logits
  k_att<<<(BB * EE + 255) / 256, 256, 0, stream>>>(SRC, DST, S1, S2, ATT);

  // 8) scatter edge scores + adjacency counts
  k_scatter<<<(BB * EE + 255) / 256, 256, 0, stream>>>(SRC, DST, ATT, SCORES, ADJ);

  // 9) masked softmax rows, times adj -> bf16 hi/lo adj_norm
  k_softmax<<<(BB * NN * 32) / 256, 256, 0, stream>>>(SCORES, ADJ, ANH, ANL);

  // 10) GEMM2 (batched): OUT[b] = leaky(adj_norm[b][N,N] @ z[b][N,D])
  {
    dim3 grid(NN / 128, DD / 64, BB);
    k_gemm_bf16x3<<<grid, 256, 0, stream>>>(ANH, ANL, ZTH, ZTL, OUT,
                                            DD, NN,
                                            (long long)NN * NN,
                                            (long long)DD * NN,
                                            (long long)NN * DD,
                                            /*leaky=*/1);
  }
}
